// CausalSelfAttention_24111946399746
// MI455X (gfx1250) — matrix operand order repaired and checked by reading, not compiled, not run on hardware
//
#include <hip/hip_runtime.h>
#include <stdint.h>

// ---------------- types ----------------
typedef __bf16 bf16_t;
typedef __attribute__((ext_vector_type(16))) __bf16 v16bf;
typedef __attribute__((ext_vector_type(8)))  float  v8f;
typedef __attribute__((ext_vector_type(4)))  int    i32x4;
typedef __attribute__((address_space(1)))    i32x4  gl_i32x4;
typedef __attribute__((address_space(3)))    i32x4  lds_i32x4;

// ---------------- problem constants ----------------
static constexpr int   Bb  = 2;
static constexpr int   T   = 2048;
static constexpr int   C   = 1024;
static constexpr int   H   = 16;
static constexpr int   HD  = 64;          // head dim
static constexpr int   N3  = 3 * C;       // 3072
static constexpr float SCALE      = 0.125f;          // 1/sqrt(64)
static constexpr float INV_KEEP_A = 1.0f / 0.9f;
static constexpr float INV_KEEP_R = 1.0f / 0.9f;

// ---------------- async global->LDS staging (gfx1250 ASYNCcnt path) -------
#define ATHENA_ASYNC __has_builtin(__builtin_amdgcn_global_load_async_to_lds_b128)

__device__ __forceinline__ void stage_b128(const bf16_t* gsrc, bf16_t* ldst) {
#if ATHENA_ASYNC
  __builtin_amdgcn_global_load_async_to_lds_b128((gl_i32x4*)gsrc, (lds_i32x4*)ldst,
                                                 0, 0);
#else
  *(uint4*)ldst = *(const uint4*)gsrc;   // synchronous fallback
#endif
}
__device__ __forceinline__ void wait_async_le1() {
#if ATHENA_ASYNC
  asm volatile("s_wait_asynccnt 0x1" ::: "memory");
#endif
}
__device__ __forceinline__ void wait_async_le0() {
#if ATHENA_ASYNC
  asm volatile("s_wait_asynccnt 0x0" ::: "memory");
#endif
}

// ---------------- fragment load helpers ----------------
// WMMA bf16 fragment layouts (CDNA5 ISA 7.12.2), contraction dim CONTIGUOUS:
//   A 16x32: lane row m = lane%16 ; K[kb..kb+7] ++ K[kb+16..kb+23], kb=(lane>=16?8:0)
//   B 32x16: lane col n = lane%16 ; K[ko..ko+7] ++ K[ko+16..ko+23], ko=(lane>=16?8:0)
union frag_u { v16bf v; uint4 q[2]; };
union pack8_u { bf16_t e[8]; uint4 q; };

__device__ __forceinline__ v16bf load_a16(const bf16_t* pr /* &row[k0] */, int lane) {
  const int kb = (lane & 16) ? 8 : 0;
  frag_u u;
  u.q[0] = *(const uint4*)(pr + kb);
  u.q[1] = *(const uint4*)(pr + kb + 16);
  return u.v;
}
__device__ __forceinline__ v16bf load_b16(const bf16_t* pc /* &colRun[k0] */, int lane) {
  const int ko = (lane & 16) ? 8 : 0;
  frag_u u;
  u.q[0] = *(const uint4*)(pc + ko);
  u.q[1] = *(const uint4*)(pc + ko + 16);
  return u.v;
}
// C/D 16x16 f32: VGPR r, lane l -> M = r + (l>=16 ? 8 : 0), N = l%16

__device__ __forceinline__ v8f wmma_bf16(const v16bf& a, const v16bf& b, const v8f& c) {
  return __builtin_amdgcn_wmma_f32_16x16x32_bf16(false, a, false, b, (short)0, c,
                                                 false, false);
}

// =====================================================================
// Kernel 0a: fp32 -> bf16, 8 contiguous elements per thread
// =====================================================================
__global__ void cvt_f32_bf16(const float* __restrict__ src, bf16_t* __restrict__ dst,
                             int n) {
  const int i = (blockIdx.x * blockDim.x + threadIdx.x) * 8;
  if (i + 8 > n) return;
  const float4 a = ((const float4*)(src + i))[0];
  const float4 b = ((const float4*)(src + i))[1];
  pack8_u u;
  u.e[0] = (bf16_t)a.x; u.e[1] = (bf16_t)a.y; u.e[2] = (bf16_t)a.z; u.e[3] = (bf16_t)a.w;
  u.e[4] = (bf16_t)b.x; u.e[5] = (bf16_t)b.y; u.e[6] = (bf16_t)b.z; u.e[7] = (bf16_t)b.w;
  *(uint4*)(dst + i) = u.q;
}

// =====================================================================
// Kernel 0b: W[K][N] fp32 -> WT[N][K] bf16 via 32x32 LDS tile (coalesced)
// =====================================================================
__global__ void transpose_cvt(const float* __restrict__ W, bf16_t* __restrict__ WT,
                              int K, int N) {
  __shared__ bf16_t tile[32][33];
  const int kt = blockIdx.x % (K / 32);
  const int nt = blockIdx.x / (K / 32);
  const int k0 = kt * 32, n0 = nt * 32;
  const int tx = threadIdx.x & 31;
  const int ty = threadIdx.x >> 5;                  // 0..7
#pragma unroll
  for (int it = 0; it < 4; ++it)
    tile[ty + 8 * it][tx] = (bf16_t)W[(size_t)(k0 + ty + 8 * it) * N + n0 + tx];
  __syncthreads();
#pragma unroll
  for (int it = 0; it < 4; ++it)
    WT[(size_t)(n0 + ty + 8 * it) * K + k0 + tx] = tile[tx][ty + 8 * it];
}

// =====================================================================
// Shared GEMM core: block = 4 waves, all on the same 32-row M-tile.
// A tile (32x32k bf16, 2 KB) double-buffered in LDS via async loads;
// each wave owns a distinct 64-col N-tile, B streamed from global (L2).
// =====================================================================
template <typename EPILOGUE>
__device__ __forceinline__ void gemm_core(const bf16_t* __restrict__ Asrc,
                                          const bf16_t* __restrict__ WT,
                                          int K, int m0, int n0, int lane,
                                          EPILOGUE&& epi) {
  __shared__ bf16_t Abuf[2][32][32];                // 2 x 2 KB
  const int colL = lane & 15;
  const int srow = threadIdx.x >> 2;                // 0..31 (128 threads)
  const int schk = (threadIdx.x & 3) * 8;

  v8f acc[2][4] = {};
  const int KSTEPS = K / 32;

  // prologue: stage k-step 0
  stage_b128(Asrc + (size_t)(m0 + srow) * K + schk, &Abuf[0][srow][schk]);

  for (int s = 0; s < KSTEPS; ++s) {
    if (s + 1 < KSTEPS) {
      stage_b128(Asrc + (size_t)(m0 + srow) * K + (s + 1) * 32 + schk,
                 &Abuf[(s + 1) & 1][srow][schk]);
      wait_async_le1();                             // step s arrived (own wave)
    } else {
      wait_async_le0();
    }
    __syncthreads();                                // all waves' stages arrived

    const bf16_t(*Ab)[32] = Abuf[s & 1];
    v16bf a[2], b[4];
    a[0] = load_a16(&Ab[colL][0], lane);
    a[1] = load_a16(&Ab[16 + colL][0], lane);
#pragma unroll
    for (int ni = 0; ni < 4; ++ni)
      b[ni] = load_b16(WT + (size_t)(n0 + ni * 16 + colL) * K + s * 32, lane);
#pragma unroll
    for (int mi = 0; mi < 2; ++mi)
#pragma unroll
      for (int ni = 0; ni < 4; ++ni)
        acc[mi][ni] = wmma_bf16(a[mi], b[ni], acc[mi][ni]);

    __syncthreads();                                // buffer reuse protection
  }
  epi(acc);
}

// =====================================================================
// Kernel 1: qkv = xbf @ WqkvT^T + bqkv ; q,k -> [B,H,T,HD], v -> [B,H,HD,T]
// =====================================================================
__global__ void __launch_bounds__(128)
qkv_gemm(const bf16_t* __restrict__ xbf, const bf16_t* __restrict__ WT,
         const float* __restrict__ bias,
         bf16_t* __restrict__ qws, bf16_t* __restrict__ kws,
         bf16_t* __restrict__ vTws) {
  const int lane  = threadIdx.x & 31;
  const int gwave = blockIdx.x * 4 + (threadIdx.x >> 5);
  const int NT = N3 / 64;                           // 48 (divisible by 4)
  const int m0 = (gwave / NT) * 32;
  const int n0 = (gwave % NT) * 64;
  const int colL = lane & 15;

  gemm_core(xbf, WT, C, m0, n0, lane, [&](v8f (&acc)[2][4]) {
#pragma unroll
    for (int mi = 0; mi < 2; ++mi)
#pragma unroll
      for (int ni = 0; ni < 4; ++ni) {
        const int n = n0 + ni * 16 + colL;
        const int which = n / C;
        const int h = (n % C) / HD;
        const int d = n % HD;
        const float bn = bias[n];
        const int mBase = m0 + mi * 16 + ((lane & 16) ? 8 : 0);
        const int bi = mBase / T;                   // tile never straddles T
        const int t0 = mBase % T;
        if (which == 2) {                           // V transposed: 1 b128 store
          pack8_u u;
#pragma unroll
          for (int r = 0; r < 8; ++r) u.e[r] = (bf16_t)(acc[mi][ni][r] + bn);
          *(uint4*)&vTws[(((size_t)bi * H + h) * HD + d) * T + t0] = u.q;
        } else {
          bf16_t* dst = (which == 0) ? qws : kws;
#pragma unroll
          for (int r = 0; r < 8; ++r)
            dst[(((size_t)bi * H + h) * T + t0 + r) * HD + d] =
                (bf16_t)(acc[mi][ni][r] + bn);
        }
      }
  });
}

// =====================================================================
// Kernel 2: causal attention for one (b,h,16-query tile) per block.
// 256 threads = 8 waves. LDS: 16 x 2048 bf16 score/prob tile (64 KB).
// =====================================================================
__global__ void attn_kernel(const bf16_t* __restrict__ qws,
                            const bf16_t* __restrict__ kws,
                            const bf16_t* __restrict__ vTws,
                            const unsigned char* __restrict__ amask,
                            bf16_t* __restrict__ yatt) {
  __shared__ bf16_t P[16][T];                       // 64 KB

  const int lane = threadIdx.x & 31;
  const int wave = threadIdx.x >> 5;                // 0..7
  const int bh = blockIdx.x / (T / 16);             // 0..B*H-1
  const int qt = blockIdx.x % (T / 16);
  const int b = bh / H, h = bh % H;
  const int q0 = qt * 16;

  const bf16_t* qbase  = qws  + (size_t)bh * T * HD;
  const bf16_t* kbase  = kws  + (size_t)bh * T * HD;
  const bf16_t* vTbase = vTws + (size_t)bh * HD * T;

  // ---- phase 0: zero the score tile (masked slots contribute 0 to P@V)
  uint4* pz = (uint4*)&P[0][0];
  for (int i = threadIdx.x; i < 16 * T * 2 / 16; i += blockDim.x)
    pz[i] = make_uint4(0u, 0u, 0u, 0u);
  __syncthreads();

  // ---- phase 1: S = (Q K^T) * scale, causal, key tiles split across 8 waves
  {
    v16bf qa[2];                                    // Q A-frags (hd 0-31, 32-63)
    const bf16_t* qr = qbase + (size_t)(q0 + (lane & 15)) * HD;
    qa[0] = load_a16(qr, lane);
    qa[1] = load_a16(qr + 32, lane);

    for (int kt = wave; kt <= qt; kt += 8) {
      const bf16_t* kr = kbase + (size_t)(kt * 16 + (lane & 15)) * HD;
      v8f s = {};
      s = wmma_bf16(qa[0], load_b16(kr, lane), s);
      s = wmma_bf16(qa[1], load_b16(kr + 32, lane), s);

      const int kk = kt * 16 + (lane & 15);         // global key of this lane
#pragma unroll
      for (int r = 0; r < 8; ++r) {
        const int q = r + ((lane & 16) ? 8 : 0);    // query within tile
        if (kk <= q0 + q) P[q][kk] = (bf16_t)(s[r] * SCALE);
      }
    }
  }
  __syncthreads();

  // ---- phase 2: row softmax + attention dropout. 16 threads per row
  //      (one half-wave per row -> shfl_xor reductions stay in-wave),
  //      4-wide key strips so the bool mask is read as uchar4.
  {
    const int row = threadIdx.x >> 4;               // 0..15
    const int sub = threadIdx.x & 15;
    const int kmax = q0 + row + 1;
    const unsigned char* mrow = amask + ((size_t)bh * T + (q0 + row)) * T;
    __builtin_prefetch(mrow + sub * 4, 0, 0);

    float mx = -1e30f;
    for (int base = sub * 4; base < kmax; base += 64)
#pragma unroll
      for (int j = 0; j < 4; ++j)
        if (base + j < kmax) mx = fmaxf(mx, (float)P[row][base + j]);
#pragma unroll
    for (int off = 8; off >= 1; off >>= 1) mx = fmaxf(mx, __shfl_xor(mx, off));

    float sum = 0.f;
    for (int base = sub * 4; base < kmax; base += 64)
#pragma unroll
      for (int j = 0; j < 4; ++j)
        if (base + j < kmax) sum += __expf((float)P[row][base + j] - mx);
#pragma unroll
    for (int off = 8; off >= 1; off >>= 1) sum += __shfl_xor(sum, off);
    const float inv = 1.0f / sum;

    for (int base = sub * 4; base < kmax; base += 64) {
      const uchar4 mm = *(const uchar4*)(mrow + base);
      const unsigned char mj[4] = {mm.x, mm.y, mm.z, mm.w};
#pragma unroll
      for (int j = 0; j < 4; ++j)
        if (base + j < kmax) {
          float p = __expf((float)P[row][base + j] - mx) * inv;
          P[row][base + j] = (bf16_t)(mj[j] ? p * INV_KEEP_A : 0.f);
        }
    }
  }
  __syncthreads();

  // ---- phase 3: O = P @ V ; 4 waves, one 16-wide hd chunk each
  if (wave < 4) {
    const int nc = wave;
    const int dcol = nc * 16 + (lane & 15);
    const bf16_t* vcol = vTbase + (size_t)dcol * T;  // contiguous along keys
    v8f o = {};
    const int nchunks = (qt + 2) >> 1;              // ceil((qt+1)/2) 32-key chunks
    for (int c = 0; c < nchunks; ++c) {
      const int kb0 = c * 32;
      const v16bf pa = load_a16(&P[lane & 15][kb0], lane);
      const v16bf vb = load_b16(vcol + kb0, lane);
      o = wmma_bf16(pa, vb, o);
    }
#pragma unroll
    for (int r = 0; r < 8; ++r) {
      const int q = r + ((lane & 16) ? 8 : 0);
      yatt[((size_t)b * T + q0 + q) * C + h * HD + dcol] = (bf16_t)o[r];
    }
  }
}

// =====================================================================
// Kernel 3: out = dropout(yatt @ WprojT^T + bproj), fp32 output
// =====================================================================
__global__ void __launch_bounds__(128)
proj_gemm(const bf16_t* __restrict__ yatt, const bf16_t* __restrict__ WT,
          const float* __restrict__ bias, const unsigned char* __restrict__ rmask,
          float* __restrict__ out) {
  const int lane  = threadIdx.x & 31;
  const int gwave = blockIdx.x * 4 + (threadIdx.x >> 5);
  const int NT = C / 64;                            // 16 (divisible by 4)
  const int m0 = (gwave / NT) * 32;
  const int n0 = (gwave % NT) * 64;
  const int colL = lane & 15;

  gemm_core(yatt, WT, C, m0, n0, lane, [&](v8f (&acc)[2][4]) {
#pragma unroll
    for (int mi = 0; mi < 2; ++mi)
#pragma unroll
      for (int ni = 0; ni < 4; ++ni) {
        const int n = n0 + ni * 16 + colL;
        const float bn = bias[n];
#pragma unroll
        for (int r = 0; r < 8; ++r) {
          const int m = m0 + mi * 16 + r + ((lane & 16) ? 8 : 0);
          const size_t idx = (size_t)m * C + n;
          const float val = acc[mi][ni][r] + bn;
          out[idx] = rmask[idx] ? val * INV_KEEP_R : 0.f;
        }
      }
  });
}

// =====================================================================
extern "C" void kernel_launch(void* const* d_in, const int* in_sizes, int n_in,
                              void* d_out, int out_size, void* d_ws, size_t ws_size,
                              hipStream_t stream) {
  (void)in_sizes; (void)n_in; (void)out_size; (void)ws_size;
  const float* x     = (const float*)d_in[0];
  const float* Wqkv  = (const float*)d_in[1];
  const float* bqkv  = (const float*)d_in[2];
  const float* Wproj = (const float*)d_in[3];
  const float* bproj = (const float*)d_in[4];
  const unsigned char* amask = (const unsigned char*)d_in[5];  // bool [B,H,T,T]
  const unsigned char* rmask = (const unsigned char*)d_in[6];  // bool [B,T,C]
  float* out = (float*)d_out;

  // workspace (bf16 elements): xbf 4M | WqkvT 3M | WprojT 1M | q 4M | k 4M
  //                            | vT 4M | yatt 4M  => 24M elems = 48 MiB
  bf16_t* xbf    = (bf16_t*)d_ws;
  bf16_t* WqkvT  = xbf    + (size_t)Bb * T * C;
  bf16_t* WprojT = WqkvT  + (size_t)C * N3;
  bf16_t* qws    = WprojT + (size_t)C * C;
  bf16_t* kws    = qws    + (size_t)Bb * H * T * HD;
  bf16_t* vTws   = kws    + (size_t)Bb * H * T * HD;
  bf16_t* yatt   = vTws   + (size_t)Bb * H * T * HD;

  const int nx = Bb * T * C;                                   // 4,194,304
  cvt_f32_bf16<<<dim3(nx / (256 * 8)), dim3(256), 0, stream>>>(x, xbf, nx);
  transpose_cvt<<<dim3((C / 32) * (N3 / 32)), dim3(256), 0, stream>>>(Wqkv, WqkvT,
                                                                      C, N3);
  transpose_cvt<<<dim3((C / 32) * (C / 32)), dim3(256), 0, stream>>>(Wproj, WprojT,
                                                                     C, C);
  qkv_gemm<<<dim3((128 * 48) / 4), dim3(128), 0, stream>>>(xbf, WqkvT, bqkv,
                                                           qws, kws, vTws);
  attn_kernel<<<dim3(Bb * H * (T / 16)), dim3(256), 0, stream>>>(qws, kws, vTws,
                                                                 amask, yatt);
  proj_gemm<<<dim3((128 * 16) / 4), dim3(128), 0, stream>>>(yatt, WprojT, bproj,
                                                            rmask, out);
}
